// VectorField_g_13520557048171
// MI455X (gfx1250) — matrix operand order repaired and checked by reading, not compiled, not run on hardware
//
#include <hip/hip_runtime.h>
#include <hip/hip_bf16.h>

// ---------------------------------------------------------------------------
// Problem constants
// ---------------------------------------------------------------------------
#define BB   64
#define NN   300
#define HH   64
#define KK   3
#define DD   10
#define NPAD 304     // 19 tiles of 16 rows for S
#define KPAD 320     // 10 k-steps of 32 for S

typedef __attribute__((ext_vector_type(16))) __bf16 v16bf;
typedef __attribute__((ext_vector_type(8)))  __bf16 v8bf;
typedef __attribute__((ext_vector_type(8)))  float  v8f;
typedef unsigned int u32x4 __attribute__((ext_vector_type(4)));
typedef int          i32x8 __attribute__((ext_vector_type(8)));
typedef int          i32x4 __attribute__((ext_vector_type(4)));

#if defined(__AMDGCN__) && __has_builtin(__builtin_amdgcn_tensor_load_to_lds) && \
    __has_builtin(__builtin_amdgcn_s_wait_tensorcnt)
#define USE_TDM 1
#else
#define USE_TDM 0
#endif

// ---------------------------------------------------------------------------
// WMMA helpers (CDNA5 v_wmma_f32_16x16x32_bf16, wave32)
// ---------------------------------------------------------------------------
__device__ __forceinline__ v8f wmma_bf16(v16bf a, v16bf b, v8f c) {
  return __builtin_amdgcn_wmma_f32_16x16x32_bf16(
      /*neg_a=*/false, a, /*neg_b=*/false, b,
      /*c_mod=*/(short)0, c, /*reuse_a=*/false, /*reuse_b=*/false);
}

// A-fragment (16x32 bf16, ISA 7.12.2): lane<16 -> M=lane, K in {0..7,16..23};
// lane>=16 -> M=lane-16, K in {8..15,24..31}.  Two contiguous 16B chunks.
// Caller passes p = &A[m][k0+klo].
__device__ __forceinline__ v16bf frag_a(const __bf16* p) {
  v8bf lo = *(const v8bf*)(p);
  v8bf hi = *(const v8bf*)(p + 16);
  v16bf r;
#pragma unroll
  for (int e = 0; e < 8; ++e) { r[e] = lo[e]; r[e + 8] = hi[e]; }
  return r;
}

// B-fragment from a pre-packed panel: each 32x16 tile is stored as
// 32 lanes x 16 bf16 contiguous (lane-major), i.e. exactly the per-lane
// register image -> one 32B vector load (2 x b128).
__device__ __forceinline__ v16bf frag_pk(const __bf16* panel, int tile, int lane) {
  return *(const v16bf*)(panel + ((long)tile * 32 + lane) * 16);
}

__device__ __forceinline__ float fast_tanh(float x) {
#if __has_builtin(__builtin_amdgcn_tanhf)
  return __builtin_amdgcn_tanhf(x);   // v_tanh_f32 (gfx1250 transcendental)
#else
  float e = __builtin_expf(-2.0f * __builtin_fabsf(x));
  float t = (1.0f - e) / (1.0f + e);
  return __builtin_copysignf(t, x);
#endif
}

// ---------------------------------------------------------------------------
// 1) supports = softmax(relu(E E^T), axis=1)      S1: f32 [300][300]
// ---------------------------------------------------------------------------
__global__ void k_supports(const float* __restrict__ E, float* __restrict__ S1) {
  int n = blockIdx.x;
  int t = threadIdx.x;                       // 512 threads
  __shared__ float En[DD];
  __shared__ float red[512];
  if (t < DD) En[t] = E[n * DD + t];
  __syncthreads();
  float dot = 0.0f;
  float v = -1e30f;
  if (t < NN) {
#pragma unroll
    for (int d = 0; d < DD; ++d) dot += En[d] * E[t * DD + d];
    dot = fmaxf(dot, 0.0f);
    v = dot;
  }
  red[t] = v;
  __syncthreads();
  for (int s = 256; s > 0; s >>= 1) {
    if (t < s) red[t] = fmaxf(red[t], red[t + s]);
    __syncthreads();
  }
  float mx = red[0];
  __syncthreads();
  float e = (t < NN) ? __builtin_expf(dot - mx) : 0.0f;
  red[t] = e;
  __syncthreads();
  for (int s = 256; s > 0; s >>= 1) {
    if (t < s) red[t] += red[t + s];
    __syncthreads();
  }
  float inv = 1.0f / red[0];
  if (t < NN) S1[n * NN + t] = e * inv;
}

// ---------------------------------------------------------------------------
// 2) Chebyshev T2 = 2*S1@S1 - I, pack S1/S2 to zero-padded bf16 [2][304][320]
// ---------------------------------------------------------------------------
__global__ void k_pack_S(const float* __restrict__ S1, __bf16* __restrict__ Sbf) {
  int idx = blockIdx.x * blockDim.x + threadIdx.x;
  if (idx >= 2 * NPAD * KPAD) return;
  int kk = idx / (NPAD * KPAD);
  int r  = idx % (NPAD * KPAD);
  int n = r / KPAD, m = r % KPAD;
  float v = 0.0f;
  if (n < NN && m < NN) {
    if (kk == 0) {
      v = S1[n * NN + m];
    } else {
      float acc = 0.0f;
      for (int t = 0; t < NN; ++t) acc += S1[n * NN + t] * S1[t * NN + m];
      v = 2.0f * acc - ((n == m) ? 1.0f : 0.0f);
    }
  }
  Sbf[idx] = (__bf16)v;
}

// ---------------------------------------------------------------------------
// 3) per-node weights, emitted directly in B-fragment panel order:
//    wpk[n] = 24 tiles (ot*6 + s), tile = 32 lanes x 16 bf16.
//    Logical B[n][krow][o], krow = k*64+i (192), o (64).
// ---------------------------------------------------------------------------
__global__ void k_weights(const float* __restrict__ E, const float* __restrict__ Wp,
                          __bf16* __restrict__ wpk) {
  int idx = blockIdx.x * blockDim.x + threadIdx.x;
  if (idx >= NN * 12288) return;
  int n = idx / 12288;
  int r = idx % 12288;
  int e = r & 15, lane = (r >> 4) & 31, tile = r >> 9;   // tile 0..23
  int s = tile % 6, ot = tile / 6;
  int krow = s * 32 + ((lane >> 4) << 3) + (e & 7) + ((e >> 3) << 4); // 0..191
  int o = ot * 16 + (lane & 15);
  int k = krow >> 6, i = krow & 63;
  float acc = 0.0f;
#pragma unroll
  for (int d = 0; d < DD; ++d)
    acc += E[n * DD + d] * Wp[(((d * KK + k) * HH + i) * HH) + o];
  wpk[idx] = (__bf16)acc;
}

__global__ void k_biasn(const float* __restrict__ E, const float* __restrict__ bp,
                        float* __restrict__ biasn) {
  int idx = blockIdx.x * blockDim.x + threadIdx.x;
  if (idx >= NN * HH) return;
  int n = idx / HH, o = idx % HH;
  float acc = 0.0f;
#pragma unroll
  for (int d = 0; d < DD; ++d) acc += E[n * DD + d] * bp[d * HH + o];
  biasn[idx] = acc;
}

// ---------------------------------------------------------------------------
// 4) W_in^T / W_out^T packed into B-fragment panels.
//    Wtb_pk: tiles (jt*2 + kt), jt 0..3, kt 0..1.
//    Wob_pk: tiles (ot*2 + kt), ot 0..255, kt 0..1.
//    Logical B[k][col] = W[col][k].
// ---------------------------------------------------------------------------
__global__ void k_pack_Win(const float* __restrict__ W, __bf16* __restrict__ Wt) {
  int idx = blockIdx.x * blockDim.x + threadIdx.x;
  if (idx >= HH * HH) return;
  int e = idx & 15, lane = (idx >> 4) & 31, tile = idx >> 9;
  int kt = tile & 1, jt = tile >> 1;
  int j = jt * 16 + (lane & 15);
  int k = kt * 32 + ((lane >> 4) << 3) + (e & 7) + ((e >> 3) << 4);
  Wt[idx] = (__bf16)W[j * HH + k];
}

__global__ void k_pack_Wout(const float* __restrict__ W, __bf16* __restrict__ Wt) {
  int idx = blockIdx.x * blockDim.x + threadIdx.x;
  if (idx >= HH * 4096) return;
  int e = idx & 15, lane = (idx >> 4) & 31, tile = idx >> 9;
  int kt = tile & 1, ot = tile >> 1;
  int o = ot * 16 + (lane & 15);
  int k = kt * 32 + ((lane >> 4) << 3) + (e & 7) + ((e >> 3) << 4);
  Wt[idx] = (__bf16)W[o * HH + k];
}

// ---------------------------------------------------------------------------
// 5) fc_in: h = relu(z @ W_in^T + b_in)  [19200,64]x[64,64] via WMMA.
//    z tile (4KB contiguous) DMA'd to LDS by the Tensor Data Mover
//    (tensor_load_to_lds + s_wait_tensorcnt), else staged manually.
//    Writes h_bf [B][N][64] and xg k=0 slot (S0 = identity).
// ---------------------------------------------------------------------------
__global__ void __launch_bounds__(128) k_fc_in(const float* __restrict__ z,
                                               const __bf16* __restrict__ Wtb_pk,
                                               const float* __restrict__ b_in,
                                               __bf16* __restrict__ h_bf,
                                               __bf16* __restrict__ xg) {
  __shared__ __align__(32) __bf16 zt[16 * 64];
  int row0 = blockIdx.x * 16;

#if USE_TDM
  __shared__ __align__(16) float zf[16 * 64];
  if (threadIdx.x < 32) {   // wave 0 issues the TDM descriptor
    unsigned long long ga = (unsigned long long)(const void*)(z + (long)row0 * HH);
    unsigned lds_addr = (unsigned)(unsigned long long)(void*)zf;
    u32x4 g0;
    g0[0] = 1u;                                            // count=1, user D#
    g0[1] = lds_addr;                                      // lds_addr[31:0]
    g0[2] = (unsigned)(ga & 0xFFFFFFFFu);                  // global_addr lo
    g0[3] = ((unsigned)(ga >> 32) & 0x1FFFFFFu) | (2u << 30); // hi + type=2
    i32x8 g1;
    g1[0] = 0x00020000;            // data_size=4B, no multicast
    g1[1] = (int)(1024u << 16);    // tensor_dim0 = 1024 (lo16 at [31:16])
    g1[2] = (int)(1u << 16);       // tensor_dim0 hi = 0, tensor_dim1 = 1
    g1[3] = (int)(1024u << 16);    // tensor_dim1 hi = 0, tile_dim0 = 1024
    g1[4] = 1;                     // tile_dim1 = 1
    g1[5] = 1024;                  // tensor_dim0_stride = 1024
    g1[6] = 0;
    g1[7] = 0;
    i32x4 gz4 = {0, 0, 0, 0};
    i32x8 gz8 = {0, 0, 0, 0, 0, 0, 0, 0};
    __builtin_amdgcn_tensor_load_to_lds(g0, g1, gz4, gz4, gz8, 0);
    __builtin_amdgcn_s_wait_tensorcnt(0);
  }
  __syncthreads();
  for (int idx = threadIdx.x; idx < 16 * 64; idx += 128)
    zt[idx] = (__bf16)zf[idx];
  __syncthreads();
#else
  for (int idx = threadIdx.x; idx < 16 * 64; idx += 128)
    zt[idx] = (__bf16)z[(row0 + (idx >> 6)) * HH + (idx & 63)];
  __syncthreads();
#endif

  int lane = threadIdx.x & 31, wave = threadIdx.x >> 5;
  int m = lane & 15;
  int klo = (lane >> 4) << 3;
  int nb = lane & 15;
  int mrow = (lane >> 4) << 3;
  int j0 = wave * 16;

  v16bf a0 = frag_a(&zt[m * 64 + 0 + klo]);
  v16bf a1 = frag_a(&zt[m * 64 + 32 + klo]);
  v16bf b0 = frag_pk(Wtb_pk, wave * 2 + 0, lane);
  v16bf b1 = frag_pk(Wtb_pk, wave * 2 + 1, lane);
  v8f c = {};
  c = wmma_bf16(a0, b0, c);
  c = wmma_bf16(a1, b1, c);

#pragma unroll
  for (int r = 0; r < 8; ++r) {
    int R = row0 + r + mrow;           // flat row = b*300 + n
    int j = j0 + nb;
    float v = fmaxf(c[r] + b_in[j], 0.0f);
    __bf16 bv = (__bf16)v;
    h_bf[R * HH + j] = bv;
    int node = R % NN, bb = R / NN;
    xg[(node * BB + bb) * (KK * HH) + j] = bv;   // k = 0 slot
  }
}

// ---------------------------------------------------------------------------
// 6) pack h into B-fragment panels per batch, K zero-padded to 320:
//    hpk[b]: 40 tiles (ct*10 + s), logical B[m][c], m 0..319 (pad 0), c 0..63.
// ---------------------------------------------------------------------------
__global__ void k_pack_h(const __bf16* __restrict__ h_bf, __bf16* __restrict__ hpk) {
  int idx = blockIdx.x * blockDim.x + threadIdx.x;
  if (idx >= BB * 20480) return;
  int b = idx / 20480;
  int r = idx % 20480;
  int e = r & 15, lane = (r >> 4) & 31, tile = r >> 9;   // tile 0..39
  int s = tile % 10, ct = tile / 10;
  int m = s * 32 + ((lane >> 4) << 3) + (e & 7) + ((e >> 3) << 4); // 0..319
  int c = ct * 16 + (lane & 15);
  hpk[idx] = (m < NN) ? h_bf[(b * NN + m) * HH + c] : (__bf16)0.0f;
}

// ---------------------------------------------------------------------------
// 7) graph_mm: xg[n][b][(kk+1)*64+c] = sum_m S_{kk+1}[n][m] * h[b][m][c]
//    per (kk,b): [304,320] x [320,64] WMMA (K padded, pads are zero).
// ---------------------------------------------------------------------------
__global__ void __launch_bounds__(128) k_graph_mm(const __bf16* __restrict__ Sbf,
                                                  const __bf16* __restrict__ hpk,
                                                  __bf16* __restrict__ xg) {
  int kk  = blockIdx.x / (BB * 19);
  int rem = blockIdx.x % (BB * 19);
  int b = rem / 19, nt = rem % 19;
  const __bf16* A  = Sbf + kk * NPAD * KPAD;     // [304][320]
  const __bf16* Bp = hpk + (long)b * 20480;      // fragment panels
  int n0 = nt * 16;

  int lane = threadIdx.x & 31, wave = threadIdx.x >> 5;  // wave = column tile
  int m = lane & 15;
  int klo = (lane >> 4) << 3;
  int nb = lane & 15;
  int mrow = (lane >> 4) << 3;

  v8f c = {};
#pragma unroll
  for (int s = 0; s < KPAD / 32; ++s) {
    v16bf a  = frag_a(A + (n0 + m) * KPAD + s * 32 + klo);
    v16bf bf = frag_pk(Bp, wave * 10 + s, lane);
    c = wmma_bf16(a, bf, c);
  }
#pragma unroll
  for (int r = 0; r < 8; ++r) {
    int n = n0 + r + mrow;
    if (n < NN) {
      int cc = wave * 16 + nb;
      xg[(n * BB + b) * (KK * HH) + (kk + 1) * HH + cc] = (__bf16)c[r];
    }
  }
}

// ---------------------------------------------------------------------------
// 8) node_mm: per node n, [64,192] x [192,64] + bias -> h2 (bf16)
// ---------------------------------------------------------------------------
__global__ void __launch_bounds__(512) k_node_mm(const __bf16* __restrict__ xg,
                                                 const __bf16* __restrict__ wpk,
                                                 const float* __restrict__ biasn,
                                                 __bf16* __restrict__ h2) {
  int n = blockIdx.x;
  const __bf16* A  = xg + (long)n * BB * (KK * HH);   // [64][192]
  const __bf16* Bp = wpk + (long)n * 12288;           // fragment panels

  int lane = threadIdx.x & 31, wave = threadIdx.x >> 5;   // 16 waves
  int bt = wave & 3, ot = wave >> 2;
  int m = lane & 15;
  int klo = (lane >> 4) << 3;
  int nb = lane & 15;
  int mrow = (lane >> 4) << 3;

  v8f c = {};
#pragma unroll
  for (int s = 0; s < (KK * HH) / 32; ++s) {
    v16bf a  = frag_a(A + (bt * 16 + m) * (KK * HH) + s * 32 + klo);
    v16bf bf = frag_pk(Bp, ot * 6 + s, lane);
    c = wmma_bf16(a, bf, c);
  }
#pragma unroll
  for (int r = 0; r < 8; ++r) {
    int bb = bt * 16 + r + mrow;
    int o = ot * 16 + nb;
    float v = c[r] + biasn[n * HH + o];
    h2[(bb * NN + n) * HH + o] = (__bf16)v;
  }
}

// ---------------------------------------------------------------------------
// 9) fc_out: out = tanh(h2 @ W_out^T + b_out)  [19200,64]x[64,4096] f32 out.
//    Memory-bound (315MB store); each block: 16 rows x 256 cols.
// ---------------------------------------------------------------------------
__global__ void __launch_bounds__(128) k_fc_out(const __bf16* __restrict__ h2,
                                                const __bf16* __restrict__ Wob_pk,
                                                const float* __restrict__ b_out,
                                                float* __restrict__ out) {
  int row0 = blockIdx.x * 16;
  const __bf16* A = h2 + (long)row0 * HH;

  int lane = threadIdx.x & 31, wave = threadIdx.x >> 5;
  int m = lane & 15;
  int klo = (lane >> 4) << 3;
  int nb = lane & 15;
  int mrow = (lane >> 4) << 3;

  v16bf a0 = frag_a(A + m * HH + 0 + klo);
  v16bf a1 = frag_a(A + m * HH + 32 + klo);

  int obase = blockIdx.y * 256 + wave * 64;
#pragma unroll
  for (int t = 0; t < 4; ++t) {
    int o0 = obase + t * 16;
    int ot = o0 >> 4;
    v16bf b0 = frag_pk(Wob_pk, ot * 2 + 0, lane);
    v16bf b1 = frag_pk(Wob_pk, ot * 2 + 1, lane);
    v8f c = {};
    c = wmma_bf16(a0, b0, c);
    c = wmma_bf16(a1, b1, c);
#pragma unroll
    for (int r = 0; r < 8; ++r) {
      long R = row0 + r + mrow;
      int o = o0 + nb;
      float v = fast_tanh(c[r] + b_out[o]);
      out[R * 4096 + o] = v;
    }
  }
}

// ---------------------------------------------------------------------------
// launch
// ---------------------------------------------------------------------------
extern "C" void kernel_launch(void* const* d_in, const int* in_sizes, int n_in,
                              void* d_out, int out_size, void* d_ws, size_t ws_size,
                              hipStream_t stream) {
  const float* z     = (const float*)d_in[0];
  const float* W_in  = (const float*)d_in[1];
  const float* b_in  = (const float*)d_in[2];
  const float* W_out = (const float*)d_in[3];
  const float* b_out = (const float*)d_in[4];
  const float* E     = (const float*)d_in[5];
  const float* Wp    = (const float*)d_in[6];
  const float* bp    = (const float*)d_in[7];
  float* out = (float*)d_out;

  char* ws = (char*)d_ws;
  float*  S1     = (float*)(ws + 0);                 // 360000 B
  __bf16* Sbf    = (__bf16*)(ws + 360192);           // 389120 B
  __bf16* Wtb_pk = (__bf16*)(ws + 749312);           // 8192 B
  __bf16* Wob_pk = (__bf16*)(ws + 757504);           // 524288 B
  __bf16* wpk    = (__bf16*)(ws + 1281792);          // 7372800 B
  float*  biasn  = (float*)(ws + 8654592);           // 76800 B
  __bf16* h_bf   = (__bf16*)(ws + 8731392);          // 2457600 B
  __bf16* hpk    = (__bf16*)(ws + 11188992);         // 2621440 B
  __bf16* xg     = (__bf16*)(ws + 13810432);         // 7372800 B
  __bf16* h2     = (__bf16*)(ws + 21183232);         // 2457600 B

  k_supports<<<NN, 512, 0, stream>>>(E, S1);
  k_pack_S<<<(2 * NPAD * KPAD + 255) / 256, 256, 0, stream>>>(S1, Sbf);
  k_weights<<<(NN * 12288 + 255) / 256, 256, 0, stream>>>(E, Wp, wpk);
  k_biasn<<<(NN * HH + 255) / 256, 256, 0, stream>>>(E, bp, biasn);
  k_pack_Win<<<(HH * HH + 255) / 256, 256, 0, stream>>>(W_in, Wtb_pk);
  k_pack_Wout<<<(HH * 4096 + 255) / 256, 256, 0, stream>>>(W_out, Wob_pk);

  k_fc_in<<<(BB * NN) / 16, 128, 0, stream>>>(z, Wtb_pk, b_in, h_bf, xg);
  k_pack_h<<<(BB * 20480 + 255) / 256, 256, 0, stream>>>(h_bf, hpk);
  k_graph_mm<<<2 * BB * 19, 128, 0, stream>>>(Sbf, hpk, xg);
  k_node_mm<<<NN, 512, 0, stream>>>(xg, wpk, biasn, h2);
  k_fc_out<<<dim3((BB * NN) / 16, 16), 128, 0, stream>>>(h2, Wob_pk, b_out, out);
}
